// TRecTransformer_31688268710093
// MI455X (gfx1250) — matrix operand order repaired and checked by reading, not compiled, not run on hardware
//
#include <hip/hip_runtime.h>
#include <hip/hip_bf16.h>
#include <cstdint>
#include <cstddef>

// ---------------------------------------------------------------------------
// Model constants (match reference)
// ---------------------------------------------------------------------------
#define D_MODEL 512
#define N_HEADS 8
#define D_HEAD  64
#define N_LAYERS 4
#define D_FF    2048
#define BATCH   4
#define L_PROJ  4096
#define L_IMG   4096
#define EPS_ATTN 1e-6f
#define EPS_LN   1e-5f
#define KV_SPLIT 8           // split-K chunks over the sequence for kv/ksum

// ---------------------------------------------------------------------------
// WMMA types / helpers (CDNA5 gfx1250, wave32)
// ---------------------------------------------------------------------------
typedef __attribute__((ext_vector_type(16))) __bf16 v16bf;
typedef __attribute__((ext_vector_type(8)))  float  v8f;

__device__ __forceinline__ __bf16 f32_to_bf16_rne(float f) {
    unsigned u = __builtin_bit_cast(unsigned, f);
    u += 0x7FFFu + ((u >> 16) & 1u);               // round-to-nearest-even
    unsigned short h = (unsigned short)(u >> 16);
    return __builtin_bit_cast(__bf16, h);
}

// Fragment-order LDS indexing: each lane's 16 bf16 elements are contiguous
// (32 bytes) so fragment loads become 2x ds_load_b128.
//
// A subtile (16x32, 16-bit): lane = (k>>4)*16 + m, slot = k&15
__device__ __forceinline__ int a_frag_idx(int m, int k) {
    int half = k >> 4;
    int e = k & 15;
    return (half * 16 + m) * 16 + e;
}
// B subtile (32x16, 16-bit): lane = (k>>4)*16 + n, slot = k&15
__device__ __forceinline__ int b_frag_idx(int k, int n) {
    return ((k >> 4) * 16 + n) * 16 + (k & 15);
}

// Activation codes for GEMM epilogue
#define ACT_NONE 0
#define ACT_RELU 1
#define ACT_ELU1 2   // elu(x)+1 : x>0 ? x+1 : exp(x)   (linear-attention feature map)

__device__ __forceinline__ float apply_act(float v, int act) {
    if (act == ACT_RELU) return fmaxf(v, 0.0f);
    if (act == ACT_ELU1) return (v > 0.0f) ? (v + 1.0f) : __expf(v);
    return v;
}

// ---------------------------------------------------------------------------
// Generic tiled WMMA GEMM:  C[M,N] = act(A[M,K] @ W[K,N] + bias[N])
// Block = 256 threads (8 waves). Tile: 128(M) x 128(N), K-step 32 (bf16).
// Wave w computes rows [16w,16w+16) x 128 cols (8 accumulators).
// LDS double-buffered; tiles stored in fragment order; B frags software-pipelined.
// ---------------------------------------------------------------------------
#define GT_M 128
#define GT_N 128
#define GT_K 32

__global__ __launch_bounds__(256)
void trt_gemm(const float* __restrict__ A, const float* __restrict__ W,
              const float* __restrict__ bias, float* __restrict__ C,
              int M, int N, int K, int act)
{
    __shared__ alignas(32) __bf16 lA[2][GT_M * GT_K];   // 8 KB per buffer
    __shared__ alignas(32) __bf16 lB[2][GT_K * GT_N];   // 8 KB per buffer

    const int tid  = threadIdx.x;
    const int wave = tid >> 5;
    const int lane = tid & 31;
    const int half = lane >> 4;
    const int l16  = lane & 15;
    const int m0   = blockIdx.y * GT_M;
    const int n0   = blockIdx.x * GT_N;

    v8f acc[8];
#pragma unroll
    for (int t = 0; t < 8; ++t) acc[t] = (v8f){0.f,0.f,0.f,0.f,0.f,0.f,0.f,0.f};

    auto stageA = [&](int k0, int buf) {
#pragma unroll
        for (int i = 0; i < 16; ++i) {                 // 128*32 = 4096 elems
            int idx = tid * 16 + i;
            int r = idx >> 5, c = idx & 31;            // row, k (coalesced reads)
            lA[buf][(r >> 4) * 512 + a_frag_idx(r & 15, c)] =
                f32_to_bf16_rne(A[(size_t)(m0 + r) * K + (k0 + c)]);
        }
    };
    auto stageB = [&](int k0, int buf) {
#pragma unroll
        for (int i = 0; i < 16; ++i) {                 // 32*128 = 4096 elems
            int idx = tid * 16 + i;
            int r = idx >> 7, c = idx & 127;           // k, n (coalesced reads)
            lB[buf][(c >> 4) * 512 + b_frag_idx(r, c & 15)] =
                f32_to_bf16_rne(W[(size_t)(k0 + r) * N + (n0 + c)]);
        }
    };

    stageA(0, 0);
    stageB(0, 0);
    __syncthreads();

    const int nk = K / GT_K;
    for (int it = 0; it < nk; ++it) {
        const int cur = it & 1;
        if (it + 1 < nk) {                             // prefetch next tile
            stageA((it + 1) * GT_K, 1 - cur);
            stageB((it + 1) * GT_K, 1 - cur);
        }
        v16bf af = *(const v16bf*)&lA[cur][wave * 512 + lane * 16];
        v16bf bc = *(const v16bf*)&lB[cur][lane * 16];
#pragma unroll
        for (int t = 0; t < 8; ++t) {                  // pipeline B frag t+1
            v16bf bn = bc;
            if (t < 7) bn = *(const v16bf*)&lB[cur][(t + 1) * 512 + lane * 16];
            acc[t] = __builtin_amdgcn_wmma_f32_16x16x32_bf16(
                false, af, false, bc, (short)0, acc[t], false, false);
            bc = bn;
        }
        __syncthreads();
    }

    // Epilogue: bias + activation. C layout: VGPR e -> row half*8+e, col l16
#pragma unroll
    for (int t = 0; t < 8; ++t) {
#pragma unroll
        for (int e = 0; e < 8; ++e) {
            int m = m0 + wave * 16 + half * 8 + e;
            int n = n0 + t * 16 + l16;
            float v = acc[t][e] + (bias ? bias[n] : 0.0f);
            C[(size_t)m * N + n] = apply_act(v, act);
        }
    }
}

// ---------------------------------------------------------------------------
// Linear attention partial: kvp[blk,d,m] = sum_{s in chunk} K[b,s,h,d]*V[b,s,h,m]
// blockIdx.x = (b*8+h)*KV_SPLIT + chunk; 128 threads = 4 waves.
// ---------------------------------------------------------------------------
__global__ __launch_bounds__(128)
void trt_kv(const float* __restrict__ Kb, const float* __restrict__ Vb,
            float* __restrict__ KVp, int Lkv)
{
    __shared__ alignas(32) __bf16 lK[2][32 * 64];   // A operand (K^T), frag order
    __shared__ alignas(32) __bf16 lV[2][32 * 64];   // B operand (V),  frag order

    const int blk = blockIdx.x;
    const int bh  = blk / KV_SPLIT, ch = blk % KV_SPLIT;
    const int b   = bh >> 3, hh = bh & 7;
    const int chunk = Lkv / KV_SPLIT;
    const int s0    = ch * chunk;
    const float* Kp = Kb + ((size_t)b * Lkv + s0) * D_MODEL + hh * D_HEAD;
    const float* Vp = Vb + ((size_t)b * Lkv + s0) * D_MODEL + hh * D_HEAD;

    const int tid = threadIdx.x, wave = tid >> 5, lane = tid & 31;
    const int half = lane >> 4, l16 = lane & 15;

    v8f acc[4];
#pragma unroll
    for (int t = 0; t < 4; ++t) acc[t] = (v8f){0.f,0.f,0.f,0.f,0.f,0.f,0.f,0.f};

    auto stage = [&](int s, int buf) {
#pragma unroll
        for (int i = 0; i < 16; ++i) {       // 2048 elems each / 128 threads
            int idx = tid * 16 + i;
            int r = idx >> 6, c = idx & 63;  // s, d|m (coalesced reads)
            // A element (m=c, k=r) of subtile c>>4 (transpose happens here)
            lK[buf][(c >> 4) * 512 + a_frag_idx(c & 15, r)] =
                f32_to_bf16_rne(Kp[(size_t)(s + r) * D_MODEL + c]);
            // B element (k=r, n=c) of subtile c>>4
            lV[buf][(c >> 4) * 512 + b_frag_idx(r, c & 15)] =
                f32_to_bf16_rne(Vp[(size_t)(s + r) * D_MODEL + c]);
        }
    };

    stage(0, 0);
    __syncthreads();

    const int nk = chunk / 32;
    for (int it = 0; it < nk; ++it) {
        const int cur = it & 1;
        if (it + 1 < nk) stage((it + 1) * 32, 1 - cur);
        v16bf af = *(const v16bf*)&lK[cur][wave * 512 + lane * 16];
        v16bf bc = *(const v16bf*)&lV[cur][lane * 16];
#pragma unroll
        for (int t = 0; t < 4; ++t) {
            v16bf bn = bc;
            if (t < 3) bn = *(const v16bf*)&lV[cur][(t + 1) * 512 + lane * 16];
            acc[t] = __builtin_amdgcn_wmma_f32_16x16x32_bf16(
                false, af, false, bc, (short)0, acc[t], false, false);
            bc = bn;
        }
        __syncthreads();
    }
#pragma unroll
    for (int t = 0; t < 4; ++t)
#pragma unroll
        for (int e = 0; e < 8; ++e) {
            int d = wave * 16 + half * 8 + e;
            int m = t * 16 + l16;
            KVp[((size_t)blk * 64 + d) * 64 + m] = acc[t][e];
        }
}

// ---------------------------------------------------------------------------
// ksum partial: ksp[blk,d] = sum_{s in chunk} K[b,s,h,d]
// blockIdx.x = (b*8+h)*KV_SPLIT + chunk; 256 threads.
// ---------------------------------------------------------------------------
__global__ __launch_bounds__(256)
void trt_colsum(const float* __restrict__ Kb, float* __restrict__ KSp, int Lkv)
{
    __shared__ float red[256];
    const int blk = blockIdx.x;
    const int bh = blk / KV_SPLIT, ch = blk % KV_SPLIT;
    const int b = bh >> 3, hh = bh & 7;
    const int chunk = Lkv / KV_SPLIT;
    const int d = threadIdx.x & 63, sub = threadIdx.x >> 6;   // 4 sub-chunks
    const float* Kp = Kb + ((size_t)b * Lkv + ch * chunk) * D_MODEL + hh * D_HEAD + d;
    const int per = chunk / 4;
    float s = 0.0f;
    for (int i = sub * per; i < (sub + 1) * per; ++i) s += Kp[(size_t)i * D_MODEL];
    red[threadIdx.x] = s;
    __syncthreads();
    if (sub == 0) KSp[blk * 64 + d] = red[d] + red[64 + d] + red[128 + d] + red[192 + d];
}

// ---------------------------------------------------------------------------
// Reduce kv / ksum partials over KV_SPLIT chunks.
// ---------------------------------------------------------------------------
__global__ void trt_kvred(const float* __restrict__ KVp, const float* __restrict__ KSp,
                          float* __restrict__ KV, float* __restrict__ KS)
{
    size_t i = (size_t)blockIdx.x * blockDim.x + threadIdx.x;
    const size_t nkv = (size_t)BATCH * N_HEADS * 64 * 64;
    if (i < nkv) {
        size_t bh = i >> 12, dm = i & 4095;
        float s = 0.0f;
#pragma unroll
        for (int c = 0; c < KV_SPLIT; ++c)
            s += KVp[((bh * KV_SPLIT + c) << 12) + dm];
        KV[i] = s;
    }
    if (i < (size_t)BATCH * N_HEADS * 64) {
        size_t bh = i >> 6, d = i & 63;
        float s = 0.0f;
#pragma unroll
        for (int c = 0; c < KV_SPLIT; ++c)
            s += KSp[(bh * KV_SPLIT + c) * 64 + d];
        KS[i] = s;
    }
}

// ---------------------------------------------------------------------------
// zinv[b,l,h] = 1 / (dot(Q[b,l,h,:], ksum[b,h,:]) + eps)
// ---------------------------------------------------------------------------
__global__ void trt_denom(const float* __restrict__ Q, const float* __restrict__ KS,
                          float* __restrict__ Zi, int Lq)
{
    size_t i = (size_t)blockIdx.x * blockDim.x + threadIdx.x;
    size_t total = (size_t)BATCH * Lq * N_HEADS;
    if (i >= total) return;
    int hh = (int)(i % N_HEADS);
    size_t bl = i / N_HEADS;
    int b = (int)(bl / Lq);
    const float* q  = Q  + bl * D_MODEL + hh * D_HEAD;
    const float* ks = KS + (size_t)(b * N_HEADS + hh) * D_HEAD;
    float s = 0.0f;
#pragma unroll 8
    for (int j = 0; j < D_HEAD; ++j) s += q[j] * ks[j];
    Zi[i] = 1.0f / (s + EPS_ATTN);
}

// ---------------------------------------------------------------------------
// attn[b,l,h*64+m] = zinv[b,l,h] * sum_d Q[b,l,h,d] * kv[b,h,d,m]
// Grid: (Lq/128, B*H). Block 256 threads; fragment-order LDS staging, K=64.
// ---------------------------------------------------------------------------
__global__ __launch_bounds__(256)
void trt_attnout(const float* __restrict__ Q, const float* __restrict__ KV,
                 const float* __restrict__ Zi, float* __restrict__ O, int Lq)
{
    __shared__ alignas(32) __bf16 lQ[128 * 64];   // [rowblk 8][kblk 2][512]
    __shared__ alignas(32) __bf16 lKV[64 * 64];   // [kblk 2][nblk 4][512]

    const int bh = blockIdx.y, b = bh >> 3, hh = bh & 7;
    const int l0 = blockIdx.x * 128;
    const float* Qp = Q + ((size_t)b * Lq + l0) * D_MODEL + hh * D_HEAD;

    const int tid = threadIdx.x, wave = tid >> 5, lane = tid & 31;
    const int half = lane >> 4, l16 = lane & 15;

#pragma unroll
    for (int i = 0; i < 16; ++i) {               // kv: 4096 elems
        int idx = tid * 16 + i;
        int r = idx >> 6, c = idx & 63;          // k, n
        lKV[(r >> 5) * 2048 + (c >> 4) * 512 + b_frag_idx(r & 31, c & 15)] =
            f32_to_bf16_rne(KV[((size_t)bh * 64 + r) * 64 + c]);
    }
#pragma unroll
    for (int i = 0; i < 32; ++i) {               // Q: 8192 elems
        int idx = tid * 32 + i;
        int r = idx >> 6, c = idx & 63;          // row, k
        lQ[(r >> 4) * 1024 + (c >> 5) * 512 + a_frag_idx(r & 15, c & 31)] =
            f32_to_bf16_rne(Qp[(size_t)r * D_MODEL + c]);
    }
    __syncthreads();

    v8f acc[4];
#pragma unroll
    for (int t = 0; t < 4; ++t) acc[t] = (v8f){0.f,0.f,0.f,0.f,0.f,0.f,0.f,0.f};

#pragma unroll
    for (int j = 0; j < 2; ++j) {                // K = 64 in two steps of 32
        v16bf af = *(const v16bf*)&lQ[wave * 1024 + j * 512 + lane * 16];
        v16bf bc = *(const v16bf*)&lKV[j * 2048 + lane * 16];
#pragma unroll
        for (int t = 0; t < 4; ++t) {
            v16bf bn = bc;
            if (t < 3) bn = *(const v16bf*)&lKV[j * 2048 + (t + 1) * 512 + lane * 16];
            acc[t] = __builtin_amdgcn_wmma_f32_16x16x32_bf16(
                false, af, false, bc, (short)0, acc[t], false, false);
            bc = bn;
        }
    }

#pragma unroll
    for (int t = 0; t < 4; ++t)
#pragma unroll
        for (int e = 0; e < 8; ++e) {
            int m = wave * 16 + half * 8 + e;          // row within 128-tile
            size_t row = (size_t)b * Lq + l0 + m;
            float z = Zi[row * N_HEADS + hh];
            O[row * D_MODEL + hh * D_HEAD + t * 16 + l16] = acc[t][e] * z;
        }
}

// ---------------------------------------------------------------------------
// LayerNorm over last dim (512), optional fused residual: y = LN(x (+ r))
// One block (256 threads) per row; GB = [gamma(512), beta(512)].
// ---------------------------------------------------------------------------
__global__ __launch_bounds__(256)
void trt_ln(const float* __restrict__ X, const float* __restrict__ R,
            const float* __restrict__ GB, float* __restrict__ Y)
{
    __shared__ float part[256];
    const size_t row = blockIdx.x;
    const int tid = threadIdx.x;
    const float* x = X + row * D_MODEL;
    float v0 = x[tid], v1 = x[tid + 256];
    if (R) { const float* r = R + row * D_MODEL; v0 += r[tid]; v1 += r[tid + 256]; }

    part[tid] = v0 + v1;
    __syncthreads();
    for (int s = 128; s > 0; s >>= 1) { if (tid < s) part[tid] += part[tid + s]; __syncthreads(); }
    float mu = part[0] * (1.0f / D_MODEL);
    __syncthreads();

    float d0 = v0 - mu, d1 = v1 - mu;
    part[tid] = d0 * d0 + d1 * d1;
    __syncthreads();
    for (int s = 128; s > 0; s >>= 1) { if (tid < s) part[tid] += part[tid + s]; __syncthreads(); }
    float rinv = rsqrtf(part[0] * (1.0f / D_MODEL) + EPS_LN);

    Y[row * D_MODEL + tid]       = d0 * rinv * GB[tid]       + GB[D_MODEL + tid];
    Y[row * D_MODEL + tid + 256] = d1 * rinv * GB[tid + 256] + GB[D_MODEL + tid + 256];
}

// ---------------------------------------------------------------------------
// Embedding: h[b,l,0:256] = x[b,l,:] @ emb_w + emb_b ; h[b,l,256:512] = pe[l,:]
// ---------------------------------------------------------------------------
__global__ void trt_embed(const float* __restrict__ x, const float* __restrict__ pe,
                          const float* __restrict__ ew, const float* __restrict__ eb,
                          float* __restrict__ H)
{
    size_t i = (size_t)blockIdx.x * blockDim.x + threadIdx.x;
    size_t total = (size_t)BATCH * L_PROJ * D_MODEL;
    if (i >= total) return;
    int c = (int)(i % D_MODEL);
    size_t bl = i / D_MODEL;
    int l = (int)(bl % L_PROJ);
    float v;
    if (c < 256) {
        const float* xp = x + bl * 2;
        v = xp[0] * ew[c] + xp[1] * ew[256 + c] + eb[c];
    } else {
        v = pe[(size_t)l * 256 + (c - 256)];
    }
    H[i] = v;
}

// Broadcast decoder queries over batch: o[b] = out_pos_emb[0]
__global__ void trt_bcast(const float* __restrict__ src, float* __restrict__ dst)
{
    size_t i = (size_t)blockIdx.x * blockDim.x + threadIdx.x;
    size_t per = (size_t)L_IMG * D_MODEL;
    if (i >= (size_t)BATCH * per) return;
    dst[i] = src[i % per];
}

// Head: out[b,l,:2] = o[b,l,:] @ pred_w + pred_b
__global__ void trt_pred(const float* __restrict__ O, const float* __restrict__ pw,
                         const float* __restrict__ pb, float* __restrict__ out)
{
    size_t i = (size_t)blockIdx.x * blockDim.x + threadIdx.x;
    if (i >= (size_t)BATCH * L_IMG) return;
    const float* o = O + i * D_MODEL;
    float s0 = pb[0], s1 = pb[1];
#pragma unroll 8
    for (int j = 0; j < D_MODEL; ++j) {
        float v = o[j];
        s0 += v * pw[j * 2];
        s1 += v * pw[j * 2 + 1];
    }
    out[i * 2]     = s0;
    out[i * 2 + 1] = s1;
}

// ---------------------------------------------------------------------------
// Host orchestration
// ---------------------------------------------------------------------------
extern "C" void kernel_launch(void* const* d_in, const int* in_sizes, int n_in,
                              void* d_out, int out_size, void* d_ws, size_t ws_size,
                              hipStream_t stream)
{
    const float* in_x        = (const float*)d_in[0];
    const float* in_ope      = (const float*)d_in[1];
    const float* in_pe       = (const float*)d_in[2];
    const float* in_embw     = (const float*)d_in[3];
    const float* in_embb     = (const float*)d_in[4];
    const float* in_eaw      = (const float*)d_in[5];
    const float* in_eab      = (const float*)d_in[6];
    const float* in_eln      = (const float*)d_in[7];
    const float* in_ew1      = (const float*)d_in[8];
    const float* in_eb1      = (const float*)d_in[9];
    const float* in_ew2      = (const float*)d_in[10];
    const float* in_eb2      = (const float*)d_in[11];
    const float* in_efln     = (const float*)d_in[12];
    const float* in_dsw      = (const float*)d_in[13];
    const float* in_dsb      = (const float*)d_in[14];
    const float* in_dcw      = (const float*)d_in[15];
    const float* in_dcb      = (const float*)d_in[16];
    const float* in_dln      = (const float*)d_in[17];
    const float* in_dw1      = (const float*)d_in[18];
    const float* in_db1      = (const float*)d_in[19];
    const float* in_dw2      = (const float*)d_in[20];
    const float* in_db2      = (const float*)d_in[21];
    const float* in_dfln     = (const float*)d_in[22];
    const float* in_pw       = (const float*)d_in[23];
    const float* in_pb       = (const float*)d_in[24];
    float* out = (float*)d_out;
    (void)in_sizes; (void)n_in; (void)out_size; (void)ws_size;

    const size_t NE = (size_t)BATCH * L_PROJ * D_MODEL;   // 8388608 (one activation)
    float* ws = (float*)d_ws;
    float* bufH   = ws;               // encoder state / scratch
    float* bufY   = ws + 1 * NE;      // attn / ffn outputs
    float* bufQ   = ws + 2 * NE;
    float* bufK   = ws + 3 * NE;
    float* bufV   = ws + 4 * NE;
    float* bufZ   = ws + 5 * NE;      // encoder memory (persists through decoder)
    float* bufO   = ws + 6 * NE;      // decoder state
    float* bufFF  = ws + 7 * NE;      // [16384, 2048] = 4*NE
    float* bufAT  = ws + 11 * NE;     // merged-head attention
    float* bufKV  = ws + 12 * NE;     // [32,64,64]
    float* bufKS  = bufKV + 32 * 64 * 64;
    float* bufZi  = bufKS + 32 * 64;                         // [B,L,H]
    float* bufKVp = bufZi + (size_t)BATCH * L_PROJ * N_HEADS; // [32*KV_SPLIT,64,64]
    float* bufKSp = bufKVp + (size_t)32 * KV_SPLIT * 64 * 64; // [32*KV_SPLIT,64]

    const int M = BATCH * L_PROJ;    // 16384 rows (same for L_IMG)

    auto gemm = [&](const float* A, const float* W, const float* bias, float* C,
                    int m, int n, int k, int act) {
        trt_gemm<<<dim3(n / GT_N, m / GT_M), 256, 0, stream>>>(A, W, bias, C, m, n, k, act);
    };

    // attention(xq, xkv) -> outbuf   (w: [4,512,512], b: [4,512])
    auto attention = [&](const float* xq, const float* xkv, const float* w, const float* b,
                         int Lq, int Lkv, float* outbuf) {
        const int Mq = BATCH * Lq, Mkv = BATCH * Lkv;
        gemm(xq,  w + 0 * D_MODEL * D_MODEL, b + 0 * D_MODEL, bufQ, Mq,  D_MODEL, D_MODEL, ACT_ELU1);
        gemm(xkv, w + 1 * D_MODEL * D_MODEL, b + 1 * D_MODEL, bufK, Mkv, D_MODEL, D_MODEL, ACT_ELU1);
        gemm(xkv, w + 2 * D_MODEL * D_MODEL, b + 2 * D_MODEL, bufV, Mkv, D_MODEL, D_MODEL, ACT_NONE);
        trt_colsum<<<BATCH * N_HEADS * KV_SPLIT, 256, 0, stream>>>(bufK, bufKSp, Lkv);
        trt_kv<<<BATCH * N_HEADS * KV_SPLIT, 128, 0, stream>>>(bufK, bufV, bufKVp, Lkv);
        {
            size_t n = (size_t)BATCH * N_HEADS * 64 * 64;
            trt_kvred<<<(unsigned)((n + 255) / 256), 256, 0, stream>>>(bufKVp, bufKSp, bufKV, bufKS);
        }
        {
            size_t n = (size_t)BATCH * Lq * N_HEADS;
            trt_denom<<<(unsigned)((n + 255) / 256), 256, 0, stream>>>(bufQ, bufKS, bufZi, Lq);
        }
        trt_attnout<<<dim3(Lq / 128, BATCH * N_HEADS), 256, 0, stream>>>(bufQ, bufKV, bufZi, bufAT, Lq);
        gemm(bufAT, w + 3 * D_MODEL * D_MODEL, b + 3 * D_MODEL, outbuf, Mq, D_MODEL, D_MODEL, ACT_NONE);
    };

    // ---- Embedding ----
    {
        size_t n = NE;
        trt_embed<<<(unsigned)((n + 255) / 256), 256, 0, stream>>>(in_x, in_pe, in_embw, in_embb, bufH);
    }

    // ---- Encoder ----
    for (int l = 0; l < N_LAYERS; ++l) {
        attention(bufH, bufH,
                  in_eaw + (size_t)l * 4 * D_MODEL * D_MODEL,
                  in_eab + (size_t)l * 4 * D_MODEL, L_PROJ, L_PROJ, bufY);
        trt_ln<<<M, 256, 0, stream>>>(bufH, bufY, in_eln + (size_t)(l * 2 + 0) * 2 * D_MODEL, bufH);
        gemm(bufH,  in_ew1 + (size_t)l * D_MODEL * D_FF, in_eb1 + (size_t)l * D_FF,
             bufFF, M, D_FF, D_MODEL, ACT_RELU);
        gemm(bufFF, in_ew2 + (size_t)l * D_FF * D_MODEL, in_eb2 + (size_t)l * D_MODEL,
             bufY, M, D_MODEL, D_FF, ACT_NONE);
        trt_ln<<<M, 256, 0, stream>>>(bufH, bufY, in_eln + (size_t)(l * 2 + 1) * 2 * D_MODEL, bufH);
    }
    trt_ln<<<M, 256, 0, stream>>>(bufH, nullptr, in_efln, bufZ);

    // ---- Decoder ----
    {
        size_t n = (size_t)BATCH * L_IMG * D_MODEL;
        trt_bcast<<<(unsigned)((n + 255) / 256), 256, 0, stream>>>(in_ope, bufO);
    }
    for (int l = 0; l < N_LAYERS; ++l) {
        attention(bufO, bufO,
                  in_dsw + (size_t)l * 4 * D_MODEL * D_MODEL,
                  in_dsb + (size_t)l * 4 * D_MODEL, L_IMG, L_IMG, bufY);
        trt_ln<<<M, 256, 0, stream>>>(bufO, bufY, in_dln + (size_t)(l * 3 + 0) * 2 * D_MODEL, bufO);
        attention(bufO, bufZ,
                  in_dcw + (size_t)l * 4 * D_MODEL * D_MODEL,
                  in_dcb + (size_t)l * 4 * D_MODEL, L_IMG, L_PROJ, bufY);
        trt_ln<<<M, 256, 0, stream>>>(bufO, bufY, in_dln + (size_t)(l * 3 + 1) * 2 * D_MODEL, bufO);
        gemm(bufO,  in_dw1 + (size_t)l * D_MODEL * D_FF, in_db1 + (size_t)l * D_FF,
             bufFF, M, D_FF, D_MODEL, ACT_RELU);
        gemm(bufFF, in_dw2 + (size_t)l * D_FF * D_MODEL, in_db2 + (size_t)l * D_MODEL,
             bufY, M, D_MODEL, D_FF, ACT_NONE);
        trt_ln<<<M, 256, 0, stream>>>(bufO, bufY, in_dln + (size_t)(l * 3 + 2) * 2 * D_MODEL, bufO);
    }
    trt_ln<<<M, 256, 0, stream>>>(bufO, nullptr, in_dfln, bufH);

    // ---- Prediction head ----
    {
        size_t n = (size_t)BATCH * L_IMG;
        trt_pred<<<(unsigned)((n + 255) / 256), 256, 0, stream>>>(bufH, in_pw, in_pb, out);
    }
}
